// M2DBlock_32985348833816
// MI455X (gfx1250) — matrix operand order repaired and checked by reading, not compiled, not run on hardware
//
#include <hip/hip_runtime.h>
#include <hip/hip_bf16.h>
#include <math.h>

// ---------------------------------------------------------------------------
// 2D selective-scan (Mamba2D) block for MI455X (gfx1250, wave32, WMMA).
// Fused design: never materialize [B,H,W,E,N] tensors (4 x 134MB in the
// reference). Workspace intermediates total ~33 MB -> L2-resident (192 MB).
// GEMMs use v_wmma_f32_16x16x32_f16 with weights pre-converted to f16 once;
// the 2D scan uses a Kogge-Stone affine scan across the 32 lanes of a wave
// (one wave per state n) with fast v_exp_f32 transcendentals.
// Requires ws_size >= ~34 MB.
// ---------------------------------------------------------------------------

typedef __attribute__((ext_vector_type(16))) _Float16 v16h;
typedef __attribute__((ext_vector_type(8)))  float    v8f;

#define PIXT 8192   // B*H*W
#define DM   128    // D_MODEL
#define EE   256    // D_INNER
#define NS   16     // D_STATE
#define RK   8      // DT_RANK
#define HH   64
#define WW   64
#define NB   2
#define DBC  48     // 2*RK + 2*NS

__device__ __forceinline__ float gelu_exact(float v) {
  return 0.5f * v * (1.0f + erff(v * 0.70710678118654752f));
}
__device__ __forceinline__ float softplus_f(float v) {
  return (v > 20.0f) ? v : log1pf(__expf(v));
}

// A-fragment, 16x32 f16 (M x K). Lane holds row M=lane&15.
// Layout: halves[t] = K = 16*hi + t (contiguous, matches B-fragment).
__device__ __forceinline__ v16h frag_a_f16(const _Float16* __restrict__ row,
                                           int kb, int hi) {
  v16h f;
#pragma unroll
  for (int t = 0; t < 16; ++t) {
    f[t] = row[kb + hi * 16 + t];
  }
  return f;
}

// B-fragment, 32x16 f16 (K x N). Lane holds column N=lane&15 = row of W
// (since B = W^T). Layout used here (same as A): halves[h] = K = 16*hi + h (contiguous 16
// halves = 32B -> two global_load_b128).
__device__ __forceinline__ v16h frag_b_f16(const _Float16* __restrict__ wrow,
                                           int kb, int hi) {
  v16h f;
#pragma unroll
  for (int h = 0; h < 16; ++h) f[h] = wrow[kb + hi * 16 + h];
  return f;
}

// ---------------------------------------------------------------------------
// Kernel 0: one-shot f32 -> f16 weight conversion (152 KB total).
// ---------------------------------------------------------------------------
__global__ __launch_bounds__(256) void k_cvtw(
    const float* __restrict__ wi, const float* __restrict__ wx,
    const float* __restrict__ wo, _Float16* __restrict__ wih,
    _Float16* __restrict__ wxh, _Float16* __restrict__ woh) {
  const int t = blockIdx.x * 256 + threadIdx.x;
  if (t < EE * DM)  wih[t] = (_Float16)wi[t];
  if (t < DBC * EE) wxh[t] = (_Float16)wx[t];
  if (t < DM * EE)  woh[t] = (_Float16)wo[t];
}

// ---------------------------------------------------------------------------
// Kernel 1: u = gelu(x @ Wi^T + bi).  Grid: 512 WGs x 512 thr (16 waves).
// Each WG: 16-pixel M-tile; wave w -> N-tile [16w,16w+16), K=128 (4 chunks).
// Writes u transposed [E][P] (f32, coalesced scan reads) + f16 row-major copy.
// ---------------------------------------------------------------------------
__global__ __launch_bounds__(512) void k_inproj(
    const float* __restrict__ x, const _Float16* __restrict__ wih,
    const float* __restrict__ bi, float* __restrict__ u_t,
    _Float16* __restrict__ uh) {
  __shared__ _Float16 xt[16][DM];  // 16 pixels x 128, f16
  const int tid  = threadIdx.x;
  const int lane = tid & 31, wave = tid >> 5;
  const int p0   = blockIdx.x * 16;

  // stage & convert the x tile once (2048 elems / 512 threads)
#pragma unroll
  for (int t = 0; t < 4; ++t) {
    int idx = tid * 4 + t;
    int m = idx >> 7, k = idx & 127;
    xt[m][k] = (_Float16)x[(size_t)(p0 + m) * DM + k];
  }
  __syncthreads();

  const int mr = lane & 15, hi = lane >> 4;
  const int n0 = wave * 16;
  const _Float16* wrow = wih + (size_t)(n0 + mr) * DM;

  v8f acc = {};
#pragma unroll
  for (int kc = 0; kc < 4; ++kc) {
    const int kb = kc * 32;
    v16h a = frag_a_f16(&xt[mr][0], kb, hi);
    v16h b = frag_b_f16(wrow, kb, hi);
    acc = __builtin_amdgcn_wmma_f32_16x16x32_f16(false, a, false, b,
                                                 (short)0, acc, false, false);
  }
  const float bias = bi[n0 + mr];
#pragma unroll
  for (int r = 0; r < 8; ++r) {
    const int pix = p0 + r + 8 * hi;   // C layout: VGPR r -> M = r + 8*hi
    const int e   = n0 + mr;           // N = lane&15
    const float g = gelu_exact(acc[r] + bias);
    u_t[(size_t)e * PIXT + pix] = g;
    uh[(size_t)pix * EE + e]    = (_Float16)g;
  }
}

// ---------------------------------------------------------------------------
// Kernel 2: dbc = u @ Wx^T + bx (48 wide, WMMA into LDS), then
// deltaT/L = softplus(d @ dtw^T + dtb) (rank-8, VALU) and B/C split.
// All outputs transposed [E][P] / [N][P] for coalesced scan reads.
// ---------------------------------------------------------------------------
__global__ __launch_bounds__(512) void k_xproj(
    const _Float16* __restrict__ uh, const _Float16* __restrict__ wxh,
    const float* __restrict__ bx, const float* __restrict__ dtTw,
    const float* __restrict__ dtTb, const float* __restrict__ dtLw,
    const float* __restrict__ dtLb, float* __restrict__ dT_t,
    float* __restrict__ dL_t, float* __restrict__ Bm_t,
    float* __restrict__ Cm_t) {
  __shared__ float dbc[16][64];  // 48 used, padded
  const int tid  = threadIdx.x;
  const int lane = tid & 31, wave = tid >> 5;
  const int p0   = blockIdx.x * 16;
  const int mr   = lane & 15, hi = lane >> 4;

  if (wave < 3) {  // wave-uniform branch: EXEC all-ones inside -> WMMA legal
    const int n0 = wave * 16;
    const _Float16* arow = uh + (size_t)(p0 + mr) * EE;
    const _Float16* brow = wxh + (size_t)(n0 + mr) * EE;
    v8f acc = {};
#pragma unroll
    for (int kc = 0; kc < 8; ++kc) {
      const int kb = kc * 32;
      v16h a = frag_a_f16(arow, kb, hi);
      v16h b = frag_b_f16(brow, kb, hi);
      acc = __builtin_amdgcn_wmma_f32_16x16x32_f16(false, a, false, b,
                                                   (short)0, acc, false, false);
    }
    const float bias = bx[n0 + mr];
#pragma unroll
    for (int r = 0; r < 8; ++r) dbc[r + 8 * hi][n0 + mr] = acc[r] + bias;
  }
  __syncthreads();

  // Per-thread: pixel m = tid/32, channels e in [8*(tid&31), +8)
  const int m   = tid >> 5;
  const int pix = p0 + m;
  const int e0  = (tid & 31) * 8;
#pragma unroll
  for (int q = 0; q < 8; ++q) {
    const int e = e0 + q;
    float sT = dtTb[e], sL = dtLb[e];
    const float* wT = dtTw + (size_t)e * RK;
    const float* wL = dtLw + (size_t)e * RK;
#pragma unroll
    for (int r = 0; r < RK; ++r) {
      sT += dbc[m][r]      * wT[r];
      sL += dbc[m][RK + r] * wL[r];
    }
    dT_t[(size_t)e * PIXT + pix] = softplus_f(sT);
    dL_t[(size_t)e * PIXT + pix] = softplus_f(sL);
  }
  const int idx = tid & 31;  // 32 values per pixel: 16 B + 16 C
  if (idx < 16) Bm_t[(size_t)idx * PIXT + pix] = dbc[m][16 + idx];
  else          Cm_t[(size_t)(idx - 16) * PIXT + pix] = dbc[m][32 + (idx - 16)];
}

// ---------------------------------------------------------------------------
// Kernel 3: 2D wavefront scan. One WG per (b,e); wave n handles state n.
// h[i,j] = exp(dT*A_T)*h[i-1,j] + [ row-scan with a=exp(dL*A_L) ] + BX.
// Row scan = affine Kogge-Stone over 32 lanes (2 columns/lane).
// Cross-n reduction in LDS (fixed order -> deterministic), fused +u*D, GELU.
// Fast __expf (v_exp_f32 TRANS, co-executes with VALU) on the serial path.
// ---------------------------------------------------------------------------
__global__ __launch_bounds__(512) void k_scan(
    const float* __restrict__ dT_t, const float* __restrict__ dL_t,
    const float* __restrict__ u_t, const float* __restrict__ Bm_t,
    const float* __restrict__ Cm_t, const float* __restrict__ ATlog,
    const float* __restrict__ ALlog, const float* __restrict__ Dv,
    _Float16* __restrict__ gy) {
  __shared__ float ypart[NS][WW + 2];
  const int tid  = threadIdx.x;
  const int lane = tid & 31, n = tid >> 5;
  const int e = blockIdx.x & (EE - 1);
  const int b = blockIdx.x >> 8;
  const size_t plane  = (size_t)e * PIXT + (size_t)b * (HH * WW);
  const size_t nplane = (size_t)n * PIXT + (size_t)b * (HH * WW);
  const float2* dTp = (const float2*)(dT_t + plane);
  const float2* dLp = (const float2*)(dL_t + plane);
  const float2* up  = (const float2*)(u_t + plane);
  const float2* bp  = (const float2*)(Bm_t + nplane);
  const float2* cp  = (const float2*)(Cm_t + nplane);

  const float aT = -expf(ATlog[e * NS + n]);
  const float aL = -expf(ALlog[e * NS + n]);
  const float dc = Dv[e];

  float h0 = 0.0f, h1 = 0.0f;  // carried state for columns 2*lane, 2*lane+1
  for (int i = 0; i < HH; ++i) {
    const int rb = i * (WW / 2);  // row base, float2 units
    const float2 dT = dTp[rb + lane];
    const float2 dL = dLp[rb + lane];
    const float2 uu = up[rb + lane];
    const float2 bm = bp[rb + lane];
    const float2 cm = cp[rb + lane];
    if (i + 1 < HH) {  // prefetch next row (global_prefetch_b8)
      __builtin_prefetch(dTp + rb + (WW / 2) + lane, 0, 1);
      __builtin_prefetch(dLp + rb + (WW / 2) + lane, 0, 1);
    }

    const float dat0 = __expf(dT.x * aT), dat1 = __expf(dT.y * aT);
    const float a0   = __expf(dL.x * aL), a1   = __expf(dL.y * aL);
    const float v0 = dat0 * h0 + (dT.x + dL.x) * bm.x * uu.x;
    const float v1 = dat1 * h1 + (dT.y + dL.y) * bm.y * uu.y;

    // combine this lane's two columns: h = As * h_left + Bs
    float As = a1 * a0;
    float Bs = a1 * v0 + v1;
    // inclusive Kogge-Stone scan of affine segments across 32 lanes
#pragma unroll
    for (int d = 1; d < 32; d <<= 1) {
      const float Ap = __shfl_up(As, d, 32);
      const float Bp = __shfl_up(Bs, d, 32);
      if (lane >= d) { Bs = As * Bp + Bs; As = As * Ap; }
    }
    float pref = __shfl_up(Bs, 1, 32);  // h at column 2*lane - 1
    if (lane == 0) pref = 0.0f;
    h0 = a0 * pref + v0;
    h1 = a1 * h0 + v1;

    ypart[n][2 * lane]     = h0 * cm.x;
    ypart[n][2 * lane + 1] = h1 * cm.y;
    __syncthreads();
    if (tid < WW) {  // waves 0-1 reduce over n in fixed order
      float s = 0.0f;
#pragma unroll
      for (int k = 0; k < NS; ++k) s += ypart[k][tid];
      const float uv = u_t[plane + (size_t)i * WW + tid];
      const float g  = gelu_exact(s + uv * dc);
      const size_t pix = (size_t)b * (HH * WW) + (size_t)i * WW + tid;
      gy[pix * EE + e] = (_Float16)g;
    }
    __syncthreads();
  }
}

// ---------------------------------------------------------------------------
// Kernel 4: out = gy @ Wo^T + bo.  Grid: 512 WGs x 256 thr (8 waves, N=128).
// ---------------------------------------------------------------------------
__global__ __launch_bounds__(256) void k_outproj(
    const _Float16* __restrict__ gy, const _Float16* __restrict__ woh,
    const float* __restrict__ bo, float* __restrict__ out) {
  const int tid  = threadIdx.x;
  const int lane = tid & 31, wave = tid >> 5;
  const int p0   = blockIdx.x * 16;
  const int mr   = lane & 15, hi = lane >> 4;
  const int n0   = wave * 16;
  const _Float16* arow = gy + (size_t)(p0 + mr) * EE;
  const _Float16* brow = woh + (size_t)(n0 + mr) * EE;

  v8f acc = {};
#pragma unroll
  for (int kc = 0; kc < 8; ++kc) {
    const int kb = kc * 32;
    v16h a = frag_a_f16(arow, kb, hi);
    v16h b = frag_b_f16(brow, kb, hi);
    acc = __builtin_amdgcn_wmma_f32_16x16x32_f16(false, a, false, b,
                                                 (short)0, acc, false, false);
  }
  const float bias = bo[n0 + mr];
#pragma unroll
  for (int r = 0; r < 8; ++r) {
    out[(size_t)(p0 + r + 8 * hi) * DM + n0 + mr] = acc[r] + bias;
  }
}

// ---------------------------------------------------------------------------
extern "C" void kernel_launch(void* const* d_in, const int* in_sizes, int n_in,
                              void* d_out, int out_size, void* d_ws,
                              size_t ws_size, hipStream_t stream) {
  const float* x    = (const float*)d_in[0];   // [P,128]
  const float* wi   = (const float*)d_in[1];   // [256,128]
  const float* bi   = (const float*)d_in[2];   // [256]
  const float* wx   = (const float*)d_in[3];   // [48,256]
  const float* bxp  = (const float*)d_in[4];   // [48]
  const float* dtTw = (const float*)d_in[5];   // [256,8]
  const float* dtTb = (const float*)d_in[6];   // [256]
  const float* dtLw = (const float*)d_in[7];   // [256,8]
  const float* dtLb = (const float*)d_in[8];   // [256]
  const float* ATl  = (const float*)d_in[9];   // [256,16]
  const float* ALl  = (const float*)d_in[10];  // [256,16]
  const float* Dvec = (const float*)d_in[11];  // [256]
  const float* wo   = (const float*)d_in[12];  // [128,256]
  const float* bo   = (const float*)d_in[13];  // [128]
  float* out = (float*)d_out;                  // [P,128]

  // Workspace carve-up (~34 MB, L2-resident)
  float*    u_t  = (float*)d_ws;                           // [E][P] f32
  _Float16* uh   = (_Float16*)(u_t + (size_t)EE * PIXT);   // [P][E] f16
  float*    dT_t = (float*)(uh + (size_t)PIXT * EE);       // [E][P] f32
  float*    dL_t = dT_t + (size_t)EE * PIXT;               // [E][P] f32
  float*    Bm_t = dL_t + (size_t)EE * PIXT;               // [N][P] f32
  float*    Cm_t = Bm_t + (size_t)NS * PIXT;               // [N][P] f32
  _Float16* gy   = (_Float16*)(Cm_t + (size_t)NS * PIXT);  // [P][E] f16
  _Float16* wih  = gy + (size_t)PIXT * EE;                 // [256][128] f16
  _Float16* wxh  = wih + (size_t)EE * DM;                  // [48][256] f16
  _Float16* woh  = wxh + (size_t)DBC * EE;                 // [128][256] f16

  const int ntiles = PIXT / 16;  // 512
  k_cvtw<<<(EE * DM + 255) / 256, 256, 0, stream>>>(wi, wx, wo, wih, wxh, woh);
  k_inproj<<<ntiles, 512, 0, stream>>>(x, wih, bi, u_t, uh);
  k_xproj<<<ntiles, 512, 0, stream>>>(uh, wxh, bxp, dtTw, dtTb, dtLw, dtLb,
                                      dT_t, dL_t, Bm_t, Cm_t);
  k_scan<<<NB * EE, 512, 0, stream>>>(dT_t, dL_t, u_t, Bm_t, Cm_t, ATl, ALl,
                                      Dvec, gy);
  k_outproj<<<ntiles, 256, 0, stream>>>(gy, woh, bo, out);
}
